// AttentionHead_21990232556069
// MI455X (gfx1250) — matrix operand order repaired and checked by reading, not compiled, not run on hardware
//
#include <hip/hip_runtime.h>
#include <hip/hip_bf16.h>

typedef __attribute__((ext_vector_type(16))) _Float16 v16h;
typedef __attribute__((ext_vector_type(8)))  _Float16 v8h;
typedef __attribute__((ext_vector_type(8)))  float    v8f;
typedef __attribute__((ext_vector_type(4)))  float    v4f;

#define NEG_INF_MASK 1.0e9f

__device__ __forceinline__ v8f wmma16x16x32(v16h a, v16h b, v8f c) {
    // D = A(16x32 f16) * B(32x16 f16) + C(16x16 f32)
    return __builtin_amdgcn_wmma_f32_16x16x32_f16(
        /*neg_a=*/false, a, /*neg_b=*/false, b,
        /*c_mod=*/(short)0, c, /*reuse_a=*/false, /*reuse_b=*/false);
}

__device__ __forceinline__ v16h cat8(v8h lo, v8h hi) {
    v16h r;
#pragma unroll
    for (int j = 0; j < 8; ++j) { r[j] = lo[j]; r[8 + j] = hi[j]; }
    return r;
}

__device__ __forceinline__ void wait_asynccnt0() {
#if __has_builtin(__builtin_amdgcn_s_wait_asynccnt)
    __builtin_amdgcn_s_wait_asynccnt(0);
#else
    asm volatile("s_wait_asynccnt 0" ::: "memory");
#endif
}

// Per-lane async copy of 16B global -> LDS (ASYNCcnt-tracked, no VGPR data).
__device__ __forceinline__ void async_g2l_b128(const void* gptr, const void* lptr) {
    const unsigned           lds = (unsigned)(size_t)lptr;  // flat low 32 = LDS byte addr
    const unsigned long long ga  = (unsigned long long)(size_t)gptr;
    asm volatile("global_load_async_to_lds_b128 %0, %1, off"
                 :: "v"(lds), "v"(ga) : "memory");
}

// ---------------------------------------------------------------------------
// Stage 1: W[D][H] f32  ->  Wt[H][D] f16   (transpose + convert)
// ---------------------------------------------------------------------------
__global__ void transpose_convert(const float* __restrict__ W,
                                  _Float16* __restrict__ Wt, int D, int H) {
    int i = blockIdx.x * blockDim.x + threadIdx.x;
    if (i >= D * H) return;
    int d = i / H;
    int h = i - d * H;
    Wt[(size_t)h * D + d] = (_Float16)W[i];
}

// ---------------------------------------------------------------------------
// Stage 2: QKV projection.  One wave per 16-row tile of S; the 4 waves of a
// block share double-buffered weight slabs staged via async global->LDS DMA.
//   qf16[B,S,H]  = (x@Wq + bq) * (1/sqrt(H))      (softmax scale folded in)
//   kf16[B,S,H]  =  x@Wk + bk
//   vt16[B,H,S]  = (x@Wv + bv)^T                  (for P@V B-operand)
// ---------------------------------------------------------------------------
#define SLAB_ELEMS (3 * 64 * 32)   // one k-step of Wq|Wk|Wv, f16

__device__ __forceinline__ void issue_slab(const _Float16* __restrict__ Wtq,
                                           const _Float16* __restrict__ Wtk,
                                           const _Float16* __restrict__ Wtv,
                                           _Float16* dst, int tid, int D, int k0) {
    // 768 x 16B transfers, 6 per thread. row = (mtx*64 + n), seg = 8-f16 chunk.
#pragma unroll
    for (int it = 0; it < 6; ++it) {
        const int t   = tid + it * 128;
        const int row = t >> 2;
        const int seg = t & 3;
        const int mtx = row >> 6;
        const int n   = row & 63;
        const _Float16* wbase = (mtx == 0) ? Wtq : (mtx == 1) ? Wtk : Wtv;
        async_g2l_b128(wbase + (size_t)n * D + k0 + seg * 8,
                       dst + row * 32 + seg * 8);
    }
}

__global__ __launch_bounds__(128)
void qkv_proj(const float* __restrict__ x,
              const _Float16* __restrict__ Wtq,
              const _Float16* __restrict__ Wtk,
              const _Float16* __restrict__ Wtv,
              const float* __restrict__ bq,
              const float* __restrict__ bk,
              const float* __restrict__ bv,
              _Float16* __restrict__ qf, _Float16* __restrict__ kf,
              _Float16* __restrict__ vt,
              int B, int S, int D, int H, float qscale) {
    __shared__ _Float16 wslab[2 * SLAB_ELEMS];   // 24 KB double buffer

    const int tid  = threadIdx.x;
    const int lane = tid & 31;
    const int wave = tid >> 5;
    const int nTiles = B * (S / 16);
    int tile = blockIdx.x * 4 + wave;
    if (tile >= nTiles) tile = nTiles - 1;       // keep barrier participation
    const int tilesPerB = S / 16;
    const int b  = tile / tilesPerB;
    const int sb = (tile - b * tilesPerB) * 16;
    const int ln = lane & 15;
    const int hi = lane >> 4;

    v8f acc[3][4];
#pragma unroll
    for (int m = 0; m < 3; ++m)
#pragma unroll
        for (int nt = 0; nt < 4; ++nt)
            acc[m][nt] = v8f{0.f, 0.f, 0.f, 0.f, 0.f, 0.f, 0.f, 0.f};

    // A-operand rows: lane holds row m=ln, k runs [8*hi,+8) and [16+8*hi,+8)
    const float* xrow = x + ((size_t)b * S + sb + ln) * D;

    issue_slab(Wtq, Wtk, Wtv, wslab, tid, D, 0);

    int step = 0;
    for (int k0 = 0; k0 < D; k0 += 32, ++step) {
        const _Float16* cur = wslab + (step & 1) * SLAB_ELEMS;
        wait_asynccnt0();          // my wave's slab DMA done
        __syncthreads();           // everyone's slab DMA done
        if (k0 + 32 < D)           // prefill other buffer while we compute
            issue_slab(Wtq, Wtk, Wtv, wslab + ((step + 1) & 1) * SLAB_ELEMS,
                       tid, D, k0 + 32);

        v4f a0 = *(const v4f*)(xrow + k0 + 8 * hi);
        v4f a1 = *(const v4f*)(xrow + k0 + 8 * hi + 4);
        v4f a2 = *(const v4f*)(xrow + k0 + 16 + 8 * hi);
        v4f a3 = *(const v4f*)(xrow + k0 + 16 + 8 * hi + 4);
        v16h A;
#pragma unroll
        for (int j = 0; j < 4; ++j) {
            A[j]      = (_Float16)a0[j];
            A[4 + j]  = (_Float16)a1[j];
            A[8 + j]  = (_Float16)a2[j];
            A[12 + j] = (_Float16)a3[j];
        }
#pragma unroll
        for (int mtx = 0; mtx < 3; ++mtx)
#pragma unroll
            for (int nt = 0; nt < 4; ++nt) {
                // B-operand from LDS: lane n = nt*16+ln, k runs [8*hi,+8) and [16+8*hi,+8)
                const _Float16* wp =
                    cur + mtx * 2048 + (nt * 16 + ln) * 32 + 8 * hi;
                v16h Bv = cat8(*(const v8h*)wp, *(const v8h*)(wp + 16));
                acc[mtx][nt] = wmma16x16x32(A, Bv, acc[mtx][nt]);
            }
        __syncthreads();           // all waves done reading before overwrite
    }

    // C layout: element (m = r + 8*hi, n = nt*16 + ln)
#pragma unroll
    for (int nt = 0; nt < 4; ++nt) {
        const int ng = nt * 16 + ln;
        const float bqv = bq[ng], bkv = bk[ng], bvv = bv[ng];
#pragma unroll
        for (int r = 0; r < 8; ++r) {
            const int mm  = r + 8 * hi;
            const size_t row = (size_t)b * S + sb + mm;
            qf[row * H + ng] = (_Float16)((acc[0][nt][r] + bqv) * qscale);
            kf[row * H + ng] = (_Float16)(acc[1][nt][r] + bkv);
            vt[((size_t)b * H + ng) * S + sb + mm] =
                (_Float16)(acc[2][nt][r] + bvv);
        }
    }
}

// ---------------------------------------------------------------------------
// Stage 3: causal flash attention.  One wave per 16-row query tile,
// 64 keys per online-softmax round (4 score tiles -> 16 WMMAs/round).
// ---------------------------------------------------------------------------
__global__ __launch_bounds__(128)
void flash_attn(const _Float16* __restrict__ qf,
                const _Float16* __restrict__ kf,
                const _Float16* __restrict__ vt,
                float* __restrict__ out, int B, int S, int H) {
    __shared__ _Float16 ldsP[4][16 * 64];     // per-wave P re-layout slab
    const int lane = threadIdx.x & 31;
    const int wave = threadIdx.x >> 5;
    _Float16* P = &ldsP[wave][0];

    const int tile = blockIdx.x * 4 + wave;
    const int tilesPerB = S / 16;
    if (tile >= B * tilesPerB) return;        // wave-uniform, no barriers here
    const int b  = tile / tilesPerB;
    const int sb = (tile - b * tilesPerB) * 16;
    const int ln = lane & 15;
    const int hi = lane >> 4;

    // Q as A-operand (K-dim = h, split into two 32-chunks)
    const _Float16* qrow = qf + ((size_t)b * S + sb + ln) * H;
    v16h qA[2];
#pragma unroll
    for (int half = 0; half < 2; ++half) {
        v8h r0 = *(const v8h*)(qrow + half * 32 + 16 * hi);
        v8h r1 = *(const v8h*)(qrow + half * 32 + 16 * hi + 8);
        qA[half] = cat8(r0, r1);
    }

    v8f acc[4];
#pragma unroll
    for (int ht = 0; ht < 4; ++ht)
        acc[ht] = v8f{0.f, 0.f, 0.f, 0.f, 0.f, 0.f, 0.f, 0.f};
    float mrow[8], lrow[8];
#pragma unroll
    for (int r = 0; r < 8; ++r) { mrow[r] = -3.0e38f; lrow[r] = 0.f; }

    const v8f zc = v8f{0.f, 0.f, 0.f, 0.f, 0.f, 0.f, 0.f, 0.f};
    const int skEnd = sb + 16;                // causal: keys < sb+16

    for (int sk0 = 0; sk0 < skEnd; sk0 += 64) {
        if (sk0 + 64 < skEnd) {               // stream next K/V block
            __builtin_prefetch(kf + ((size_t)b * S + sk0 + 64 + ln) * H, 0, 1);
            __builtin_prefetch(vt + ((size_t)b * H + ln) * S + sk0 + 64, 0, 1);
        }
        // ---- scores for four 16-key blocks: S = Q(16x64) @ K^T
        v8f s[4];
#pragma unroll
        for (int nt = 0; nt < 4; ++nt) {
            const _Float16* krow =
                kf + ((size_t)b * S + sk0 + nt * 16 + ln) * H;
            v16h kB0 = cat8(*(const v8h*)(krow + 16 * hi),
                            *(const v8h*)(krow + 16 * hi + 8));
            v16h kB1 = cat8(*(const v8h*)(krow + 32 + 16 * hi),
                            *(const v8h*)(krow + 32 + 16 * hi + 8));
            v8f sc = wmma16x16x32(qA[0], kB0, zc);
            sc     = wmma16x16x32(qA[1], kB1, sc);
            s[nt] = sc;
        }
        // ---- causal mask + per-lane block max
        float bmax[8];
#pragma unroll
        for (int r = 0; r < 8; ++r) {
            const int sq = sb + r + 8 * hi;
            float mx = -3.0e38f;
#pragma unroll
            for (int nt = 0; nt < 4; ++nt) {
                float sv = s[nt][r] -
                    ((sq < sk0 + nt * 16 + ln) ? NEG_INF_MASK : 0.f);
                s[nt][r] = sv;
                mx = fmaxf(mx, sv);
            }
            bmax[r] = mx;
        }
        // row max: reduce across each 16-lane half (rows r / r+8)
#pragma unroll
        for (int off = 1; off < 16; off <<= 1)
#pragma unroll
            for (int r = 0; r < 8; ++r)
                bmax[r] = fmaxf(bmax[r], __shfl_xor(bmax[r], off, 32));

        float corr[8], psum[8];
#pragma unroll
        for (int r = 0; r < 8; ++r) {
            const float nm = fmaxf(mrow[r], bmax[r]);
            corr[r] = __expf(mrow[r] - nm);
            mrow[r] = nm;
            float ps = 0.f;
#pragma unroll
            for (int nt = 0; nt < 4; ++nt) {
                const float p = __expf(s[nt][r] - nm);
                s[nt][r] = p;
                ps += p;
            }
            psum[r] = ps;
        }
#pragma unroll
        for (int off = 1; off < 16; off <<= 1)
#pragma unroll
            for (int r = 0; r < 8; ++r)
                psum[r] += __shfl_xor(psum[r], off, 32);
#pragma unroll
        for (int r = 0; r < 8; ++r) lrow[r] = lrow[r] * corr[r] + psum[r];
#pragma unroll
        for (int ht = 0; ht < 4; ++ht)
#pragma unroll
            for (int r = 0; r < 8; ++r) acc[ht][r] *= corr[r];

        // ---- re-layout P (C layout -> A layout) through LDS
#pragma unroll
        for (int nt = 0; nt < 4; ++nt)
#pragma unroll
            for (int r = 0; r < 8; ++r)
                P[(r + 8 * hi) * 64 + nt * 16 + ln] = (_Float16)s[nt][r];
        v16h pA0 = cat8(*(const v8h*)(P + ln * 64 + 16 * hi),
                        *(const v8h*)(P + ln * 64 + 16 * hi + 8));
        v16h pA1 = cat8(*(const v8h*)(P + ln * 64 + 32 + 16 * hi),
                        *(const v8h*)(P + ln * 64 + 40 + 16 * hi));

        // ---- acc += P(16x64) @ V(64x16) for each 16-wide h tile
#pragma unroll
        for (int ht = 0; ht < 4; ++ht) {
            const _Float16* vrow =
                vt + ((size_t)b * H + ht * 16 + ln) * S + sk0;
            v16h vB0 = cat8(*(const v8h*)(vrow + 16 * hi),
                            *(const v8h*)(vrow + 16 * hi + 8));
            v16h vB1 = cat8(*(const v8h*)(vrow + 32 + 16 * hi),
                            *(const v8h*)(vrow + 32 + 16 * hi + 8));
            acc[ht] = wmma16x16x32(pA0, vB0, acc[ht]);
            acc[ht] = wmma16x16x32(pA1, vB1, acc[ht]);
        }
    }

    // ---- normalize and store f32 output [B,S,H]
#pragma unroll
    for (int ht = 0; ht < 4; ++ht)
#pragma unroll
        for (int r = 0; r < 8; ++r) {
            const int mm = r + 8 * hi;
            out[((size_t)b * S + sb + mm) * H + ht * 16 + ln] =
                acc[ht][r] / lrow[r];
        }
}

// ---------------------------------------------------------------------------
extern "C" void kernel_launch(void* const* d_in, const int* in_sizes, int n_in,
                              void* d_out, int out_size, void* d_ws,
                              size_t ws_size, hipStream_t stream) {
    (void)n_in; (void)out_size; (void)ws_size;
    const float* x  = (const float*)d_in[0];
    const float* Wq = (const float*)d_in[1];
    const float* bq = (const float*)d_in[2];
    const float* Wk = (const float*)d_in[3];
    const float* bk = (const float*)d_in[4];
    const float* Wv = (const float*)d_in[5];
    const float* bv = (const float*)d_in[6];

    const int S = 4096;
    const int H = in_sizes[2];              // 64
    const int D = in_sizes[1] / H;          // 1024
    const int B = in_sizes[0] / (S * D);    // 4

    // workspace layout (all offsets 256B-aligned)
    char* ws = (char*)d_ws;
    const size_t wtBytes = (size_t)H * D * sizeof(_Float16);     // 128 KB
    const size_t qkBytes = (size_t)B * S * H * sizeof(_Float16); // 2 MB
    _Float16* Wtq = (_Float16*)(ws);
    _Float16* Wtk = (_Float16*)(ws + wtBytes);
    _Float16* Wtv = (_Float16*)(ws + 2 * wtBytes);
    _Float16* qf  = (_Float16*)(ws + 3 * wtBytes);
    _Float16* kf  = (_Float16*)(ws + 3 * wtBytes + qkBytes);
    _Float16* vt  = (_Float16*)(ws + 3 * wtBytes + 2 * qkBytes);

    const int nW = D * H;
    transpose_convert<<<(nW + 255) / 256, 256, 0, stream>>>(Wq, Wtq, D, H);
    transpose_convert<<<(nW + 255) / 256, 256, 0, stream>>>(Wk, Wtk, D, H);
    transpose_convert<<<(nW + 255) / 256, 256, 0, stream>>>(Wv, Wtv, D, H);

    const int tiles  = (B * S) / 16;        // 1024 waves, 4 waves/block
    const int blocks = (tiles + 3) / 4;
    const float qscale = 1.0f / sqrtf((float)H);
    qkv_proj<<<blocks, 128, 0, stream>>>(x, Wtq, Wtk, Wtv, bq, bk, bv,
                                         qf, kf, vt, B, S, D, H, qscale);
    flash_attn<<<blocks, 128, 0, stream>>>(qf, kf, vt, (float*)d_out, B, S, H);
}
